// LinearGaussianChain_21036749816041
// MI455X (gfx1250) — matrix operand order repaired and checked by reading, not compiled, not run on hardware
//
#include <hip/hip_runtime.h>
#include <math.h>

#define NODE 64
#define TT   256
#define MB   2048
#define XS   65   // padded LDS row stride (floats) -> (row + k) bank pattern

typedef __attribute__((ext_vector_type(16))) _Float16 v16h;
typedef __attribute__((ext_vector_type(8)))  float    v8f;
typedef __attribute__((ext_vector_type(4)))  float    v4f;

__device__ __forceinline__ float softplus_f(float v) {
    return (v > 20.0f) ? v : log1pf(expf(v));
}

// Pack W[t]^T (i.e. B[k][n] = W[n][k]) into per-lane WMMA B-fragment order,
// split into hi/lo f16 halves (Markidis split -> near-fp32 product accuracy).
// Fragment (t, nt, kb): 32 lanes x 16 halfs, flat idx =
//   t*4096 + nt*1024 + kb*512 + lane*16 + e
// lane = hi*16 + ll holds column n = nt*16 + ll, K = kb*32 + e + 16*hi.
__global__ __launch_bounds__(256) void prep_w(const float* __restrict__ W,
                                              _Float16* __restrict__ Wh,
                                              _Float16* __restrict__ Wl) {
    int idx  = blockIdx.x * 256 + threadIdx.x;   // 0 .. TT*4096-1
    int e    = idx & 15;
    int lane = (idx >> 4) & 31;
    int kb   = (idx >> 9) & 1;
    int nt   = (idx >> 10) & 3;
    int t    = idx >> 12;
    int ll = lane & 15, hi = lane >> 4;
    int n  = nt * 16 + ll;
    int k  = kb * 32 + e + 16 * hi;
    float w = W[t * 4096 + n * 64 + k];          // B[k][n] = W[n][k]
    _Float16 wh = (_Float16)w;
    Wh[idx] = wh;
    Wl[idx] = (_Float16)(w - (float)wh);
}

// One wave32 per block carries 16 batch rows through all 255 steps.
// State x_{t-1} lives in LDS (f32); outputs staged 4 t-slices deep in LDS
// and flushed as reversed float4 runs for 16B-granular coalesced stores.
__global__ __launch_bounds__(32) void chain(
    const float* __restrict__ bvec,   // (T, 64)
    const float* __restrict__ ne,     // (1, 64, T)
    const float* __restrict__ eps0,   // (M, 64)
    const float* __restrict__ eps,    // (T-1, M, 64)
    const _Float16* __restrict__ Wh,
    const _Float16* __restrict__ Wl,
    float* __restrict__ out_x,
    float* __restrict__ out_m,
    float* __restrict__ out_s)
{
    __shared__ float xbuf[16 * XS];          // 4.2 KB recurrence state
    __shared__ float stage[3][16 * 64 * 4];  // 48 KB: x/m/s, 4 t-slots

    const int lane = threadIdx.x;
    const int ll = lane & 15, hi = lane >> 4;
    const int row0 = blockIdx.x * 16;

    // ---- t = 0: x0 = eps0, m0 = 0, s0 = IN_SCALE ----
    #pragma unroll 4
    for (int it = 0; it < 32; ++it) {
        int i = it * 32 + lane;              // i = rl*64 + n
        float x0 = eps0[row0 * 64 + i];
        int rl = i >> 6, n = i & 63;
        xbuf[rl * XS + n] = x0;
        stage[0][i * 4 + 0] = x0;
        stage[1][i * 4 + 0] = 0.0f;
        stage[2][i * 4 + 0] = 1.0f;
    }
    __syncthreads();   // 1-wave WG: lowers to a cheap fence

    for (int t = 1; t < TT; ++t) {
        // per-column bias and noise scale (n = nt*16 + ll)
        float bv[4], sv[4];
        #pragma unroll
        for (int nt = 0; nt < 4; ++nt) {
            int n = nt * 16 + ll;
            bv[nt] = bvec[t * 64 + n];
            sv[nt] = softplus_f(ne[n * 256 + t]);
        }

        // A fragments (x_{t-1}) from LDS: contiguous A layout k = e + 16*hi,
        // f32 -> hi/lo f16 split
        v16h ah0, al0, ah1, al1;
        #pragma unroll
        for (int e = 0; e < 16; ++e) {
            int k0 = e + 16 * hi;
            float a0 = xbuf[ll * XS + k0];
            float a1 = xbuf[ll * XS + 32 + k0];
            _Float16 h0 = (_Float16)a0;
            _Float16 h1 = (_Float16)a1;
            ((_Float16*)&ah0)[e] = h0;
            ((_Float16*)&al0)[e] = (_Float16)(a0 - (float)h0);
            ((_Float16*)&ah1)[e] = h1;
            ((_Float16*)&al1)[e] = (_Float16)(a1 - (float)h1);
        }

        const _Float16* whp = Wh + (size_t)t * 4096 + lane * 16;
        const _Float16* wlp = Wl + (size_t)t * 4096 + lane * 16;
        if (t + 1 < TT) {                    // pull next step's W into cache
            __builtin_prefetch(whp + 4096, 0, 2);
            __builtin_prefetch(wlp + 4096, 0, 2);
        }

        // 4 N-tiles x (2 K-blocks x 3 split terms) = 24 WMMAs,
        // 8 independent accumulation chains (p/q per tile) for ILP
        v8f p[4], q[4];
        #pragma unroll
        for (int nt = 0; nt < 4; ++nt) {
            v16h bh0 = *(const v16h*)(whp + nt * 1024);
            v16h bh1 = *(const v16h*)(whp + nt * 1024 + 512);
            v16h bl0 = *(const v16h*)(wlp + nt * 1024);
            v16h bl1 = *(const v16h*)(wlp + nt * 1024 + 512);
            v8f pp = {}, qq = {};
            pp = __builtin_amdgcn_wmma_f32_16x16x32_f16(false, ah0, false, bh0, (short)0, pp, false, false);
            qq = __builtin_amdgcn_wmma_f32_16x16x32_f16(false, al0, false, bh0, (short)0, qq, false, false);
            qq = __builtin_amdgcn_wmma_f32_16x16x32_f16(false, ah0, false, bl0, (short)0, qq, false, false);
            pp = __builtin_amdgcn_wmma_f32_16x16x32_f16(false, ah1, false, bh1, (short)0, pp, false, false);
            qq = __builtin_amdgcn_wmma_f32_16x16x32_f16(false, al1, false, bh1, (short)0, qq, false, false);
            qq = __builtin_amdgcn_wmma_f32_16x16x32_f16(false, ah1, false, bl1, (short)0, qq, false, false);
            p[nt] = pp; q[nt] = qq;
        }

        // epilogue: m_t = D + b ; x_t = m_t + s*eps ; stage outputs, renew state
        const float* ep = eps + ((size_t)(t - 1) * MB + row0) * 64;
        const int slot = t & 3;
        #pragma unroll
        for (int nt = 0; nt < 4; ++nt) {
            int n = nt * 16 + ll;            // D layout: lane = n, vgpr = m
            #pragma unroll
            for (int v = 0; v < 8; ++v) {
                int rl = v + 8 * hi;
                float mv = p[nt][v] + q[nt][v] + bv[nt];
                float xv = mv + sv[nt] * ep[rl * 64 + n];
                int si = (rl * 64 + n) * 4 + slot;
                stage[0][si] = xv;
                stage[1][si] = mv;
                stage[2][si] = sv[nt];
                xbuf[rl * XS + n] = xv;
            }
        }

        // flush 4 t-slices as reversed float4 (t_out = 255 - t runs backwards)
        if (slot == 3) {
            __syncthreads();
            int tb = 255 - t;                // t_out of slot 3 (lowest addr)
            #pragma unroll 4
            for (int it = 0; it < 32; ++it) {
                int i = it * 32 + lane;
                int rl = i >> 6, n = i & 63;
                size_t o = ((size_t)((row0 + rl) * 64 + n)) * 256 + tb;
                const float* sx = &stage[0][i * 4];
                const float* sm = &stage[1][i * 4];
                const float* ss = &stage[2][i * 4];
                *(v4f*)(out_x + o) = (v4f){sx[3], sx[2], sx[1], sx[0]};
                *(v4f*)(out_m + o) = (v4f){sm[3], sm[2], sm[1], sm[0]};
                *(v4f*)(out_s + o) = (v4f){ss[3], ss[2], ss[1], ss[0]};
            }
        }
        __syncthreads();
    }
}

extern "C" void kernel_launch(void* const* d_in, const int* in_sizes, int n_in,
                              void* d_out, int out_size, void* d_ws, size_t ws_size,
                              hipStream_t stream) {
    (void)in_sizes; (void)n_in; (void)out_size; (void)ws_size;
    const float* W    = (const float*)d_in[0];
    const float* b    = (const float*)d_in[1];
    const float* ne   = (const float*)d_in[2];
    const float* eps0 = (const float*)d_in[3];
    const float* eps  = (const float*)d_in[4];

    _Float16* Wh = (_Float16*)d_ws;                 // 2 MB
    _Float16* Wl = Wh + (size_t)TT * 4096;          // 2 MB

    float* out_x = (float*)d_out;
    float* out_m = out_x + (size_t)MB * NODE * TT;
    float* out_s = out_m + (size_t)MB * NODE * TT;

    prep_w<<<(TT * 4096) / 256, 256, 0, stream>>>(W, Wh, Wl);
    chain<<<MB / 16, 32, 0, stream>>>(b, ne, eps0, eps, Wh, Wl,
                                      out_x, out_m, out_s);
}
